// RevSelfScaledDotProductAttention_65223373357267
// MI455X (gfx1250) — compile-verified, matrix operand order checked by reading, not run on hardware
//
#include <hip/hip_runtime.h>
#include <hip/hip_bf16.h>

typedef _Float16 half8 __attribute__((ext_vector_type(8)));
typedef _Float16 v16h  __attribute__((ext_vector_type(16)));
typedef float    v8f   __attribute__((ext_vector_type(8)));
typedef int      v4i   __attribute__((ext_vector_type(4)));

constexpr int   BATCH = 8;
constexpr int   SEQ   = 4096;
constexpr int   DIM   = 512;
constexpr float TEMP_INV = 1.0f / 22.627416997969522f;   // 1/sqrt(512)

// ---------------------------------------------------------------------------
// CDNA5 async global->LDS copy (LDS DMA, ASYNCcnt-tracked), with fallback.
// Probe result: builtin exists; arg0 = v4i addrspace(1)* (global src),
// arg1 = v4i addrspace(3)* (LDS dst), then imm offset, imm cpol.
// ---------------------------------------------------------------------------
#if defined(__has_builtin)
#if __has_builtin(__builtin_amdgcn_global_load_async_to_lds_b128)
#define HAVE_ASYNC_LDS 1
#endif
#endif
#ifndef HAVE_ASYNC_LDS
#define HAVE_ASYNC_LDS 0
#endif

__device__ __forceinline__ void copy16_async(_Float16* lds, const _Float16* gsrc)
{
#if HAVE_ASYNC_LDS
    __builtin_amdgcn_global_load_async_to_lds_b128(
        (__attribute__((address_space(1))) v4i*)gsrc,
        (__attribute__((address_space(3))) v4i*)lds,
        0 /*offset*/, 0 /*cpol*/);
#else
    *(half8*)lds = *(const half8*)gsrc;
#endif
}

__device__ __forceinline__ void wait_async_le4()
{
#if HAVE_ASYNC_LDS
#if __has_builtin(__builtin_amdgcn_s_wait_asynccnt)
    __builtin_amdgcn_s_wait_asynccnt(4);
#else
    asm volatile("s_wait_asynccnt 4" ::: "memory");
#endif
#endif
}

__device__ __forceinline__ void wait_async_le0()
{
#if HAVE_ASYNC_LDS
#if __has_builtin(__builtin_amdgcn_s_wait_asynccnt)
    __builtin_amdgcn_s_wait_asynccnt(0);
#else
    asm volatile("s_wait_asynccnt 0" ::: "memory");
#endif
#endif
}

// ---------------------------------------------------------------------------
// fp32 -> f16 convert + 32x32 LDS transpose.
//   in       : [B][SEQ][DIM] fp32
//   outT     : [B][DIM][SEQ] f16   (transposed, scaled)
//   outPlain : [B][SEQ][DIM] f16   (optional plain copy, scaled)
// ---------------------------------------------------------------------------
__global__ __launch_bounds__(256)
void cvt_tr_kernel(const float* __restrict__ in,
                   _Float16* __restrict__ outT,
                   _Float16* __restrict__ outPlain,
                   float scale)
{
    __shared__ float tile[32][33];
    const int  b    = blockIdx.z;
    const int  n0   = blockIdx.x * 32;
    const int  d0   = blockIdx.y * 32;
    const long base = (long)b * SEQ * DIM;
    const int  tid  = threadIdx.x;
    const int  c    = tid & 31;
    const int  r0   = tid >> 5;

    for (int rr = r0; rr < 32; rr += 8) {
        float x = in[base + (long)(n0 + rr) * DIM + d0 + c] * scale;
        tile[rr][c] = x;
        if (outPlain)
            outPlain[base + (long)(n0 + rr) * DIM + d0 + c] = (_Float16)x;
    }
    __syncthreads();
    for (int rr = r0; rr < 32; rr += 8) {
        outT[base + (long)(d0 + rr) * SEQ + n0 + c] = (_Float16)tile[c][rr];
    }
}

// ---------------------------------------------------------------------------
// Tiled WMMA GEMM:  C[M][N] = A[M][K] * Bt[N][K]^T   (f16 in, f32 accumulate)
// Double-buffered LDS; async LDS-DMA staging overlapped with WMMA compute.
// ---------------------------------------------------------------------------
template <int BM, int BN, int WM, int WN, int NWAVES>
__global__ __launch_bounds__(NWAVES * 32)
void wmma_gemm_kernel(const _Float16* __restrict__ A,   // [M][K] row-major
                      const _Float16* __restrict__ Bt,  // [N][K] row-major (B^T)
                      float* __restrict__ C,            // [M][ldc]
                      int K, int ldc,
                      long strideA, long strideB, long strideC)
{
    constexpr int KT         = 32;
    constexpr int LDS_STRIDE = KT + 8;       // 40 halves = 80 B row pitch
    constexpr int WROWS      = BM / WM;
    constexpr int FM         = WM / 16;
    constexpr int FN         = WN / 16;
    constexpr int NT         = NWAVES * 32;
    constexpr int CA         = (BM * 4) / NT;   // b128 chunks per thread (A)
    constexpr int CB         = (BN * 4) / NT;   // b128 chunks per thread (B)
    static_assert(CA + CB == 4, "wait_async_le4 assumes 4 in-flight DMAs/wave");

    __shared__ alignas(16) _Float16 sA[2][BM][LDS_STRIDE];
    __shared__ alignas(16) _Float16 sB[2][BN][LDS_STRIDE];

    const int tid   = threadIdx.x;
    const int lane  = tid & 31;
    const int wave  = tid >> 5;
    const int wm    = wave % WROWS;
    const int wn    = wave / WROWS;
    const int mlane = lane & 15;             // fragment row (M or N)
    const int khalf = lane >> 4;             // which K-half this lane owns

    const _Float16* a = A  + blockIdx.z * strideA + (long)(blockIdx.x * BM) * K;
    const _Float16* b = Bt + blockIdx.z * strideB + (long)(blockIdx.y * BN) * K;

    auto stage = [&](int bufsel, int k0) {
        #pragma unroll
        for (int u = 0; u < CA; ++u) {
            const int idx = tid + u * NT;
            const int r = idx >> 2, cc = idx & 3;
            copy16_async(&sA[bufsel][r][cc * 8], a + (long)r * K + k0 + cc * 8);
        }
        #pragma unroll
        for (int u = 0; u < CB; ++u) {
            const int idx = tid + u * NT;
            const int r = idx >> 2, cc = idx & 3;
            copy16_async(&sB[bufsel][r][cc * 8], b + (long)r * K + k0 + cc * 8);
        }
    };

    v8f acc[FM][FN] = {};

    stage(0, 0);                             // prologue: tile 0 in flight
    int buf = 0;
    for (int k0 = 0; k0 < K; k0 += KT) {
        if (k0 + KT < K) {
            stage(buf ^ 1, k0 + KT);         // overlap next tile's DMA
            wait_async_le4();                // current tile's 4 DMAs retired
        } else {
            wait_async_le0();
        }
        __syncthreads();                     // all waves' DMAs visible

        // LDS -> fragments: ISA 16-bit A layout, two b128 loads per fragment
        v16h af[FM], bf[FN];
        #pragma unroll
        for (int i = 0; i < FM; ++i) {
            const _Float16* p = &sA[buf][wm * WM + i * 16 + mlane][khalf * 8];
            half8 lo = *(const half8*)p;          // K = 8*khalf .. +7
            half8 hi = *(const half8*)(p + 16);   // K = 16 + 8*khalf .. +7
            #pragma unroll
            for (int q = 0; q < 8; ++q) { af[i][q] = lo[q]; af[i][q + 8] = hi[q]; }
        }
        #pragma unroll
        for (int j = 0; j < FN; ++j) {
            const _Float16* p = &sB[buf][wn * WN + j * 16 + mlane][khalf * 8];
            half8 lo = *(const half8*)p;
            half8 hi = *(const half8*)(p + 16);
            #pragma unroll
            for (int q = 0; q < 8; ++q) { bf[j][q] = lo[q]; bf[j][q + 8] = hi[q]; }
        }

        #pragma unroll
        for (int i = 0; i < FM; ++i)
            #pragma unroll
            for (int j = 0; j < FN; ++j)
                acc[i][j] = __builtin_amdgcn_wmma_f32_16x16x32_f16(
                    false, af[i], false, bf[j], (short)0, acc[i][j], false, false);

        __syncthreads();                     // reads done before buffer reuse
        buf ^= 1;
    }

    // C/D fragment layout: elem r -> row = 8*(lane>>4)+r, col = lane&15
    float* cpt = C + blockIdx.z * strideC;
    const int ccol0 = blockIdx.y * BN + wn * WN + (lane & 15);
    #pragma unroll
    for (int i = 0; i < FM; ++i) {
        const int rbase = blockIdx.x * BM + wm * WM + i * 16 + (lane >> 4) * 8;
        #pragma unroll
        for (int j = 0; j < FN; ++j)
            #pragma unroll
            for (int r = 0; r < 8; ++r)
                cpt[(long)(rbase + r) * ldc + ccol0 + j * 16] = acc[i][j][r];
    }
}

// ---------------------------------------------------------------------------
// Row softmax over D=512; writes fp32 attn (to d_out) and f16 attn^T (ws).
// ---------------------------------------------------------------------------
__global__ __launch_bounds__(256)
void softmax_kernel(const float* __restrict__ S,
                    float* __restrict__ attn,
                    _Float16* __restrict__ attnT)
{
    const int D   = DIM;
    const int b   = blockIdx.y;
    const int row = blockIdx.x;
    const float* s = S + ((long)b * D + row) * D;
    const int tid = threadIdx.x;

    __shared__ float red[8];

    float x0 = s[tid], x1 = s[tid + 256];
    float m = fmaxf(x0, x1);
    #pragma unroll
    for (int off = 16; off > 0; off >>= 1) m = fmaxf(m, __shfl_xor(m, off));
    if ((tid & 31) == 0) red[tid >> 5] = m;
    __syncthreads();
    float mm = red[0];
    #pragma unroll
    for (int i = 1; i < 8; ++i) mm = fmaxf(mm, red[i]);
    __syncthreads();

    float e0 = expf(x0 - mm), e1 = expf(x1 - mm);
    float t = e0 + e1;
    #pragma unroll
    for (int off = 16; off > 0; off >>= 1) t += __shfl_xor(t, off);
    if ((tid & 31) == 0) red[tid >> 5] = t;
    __syncthreads();
    float sum = 0.f;
    #pragma unroll
    for (int i = 0; i < 8; ++i) sum += red[i];
    const float inv = 1.0f / sum;

    const float a0 = e0 * inv, a1 = e1 * inv;
    float* arow = attn + ((long)b * D + row) * D;
    arow[tid]       = a0;
    arow[tid + 256] = a1;
    _Float16* at = attnT + (long)b * D * D + row;   // attnT[b][e][d]
    at[(long)tid * D]         = (_Float16)a0;
    at[(long)(tid + 256) * D] = (_Float16)a1;
}

// ---------------------------------------------------------------------------
extern "C" void kernel_launch(void* const* d_in, const int* in_sizes, int n_in,
                              void* d_out, int out_size, void* d_ws, size_t ws_size,
                              hipStream_t stream)
{
    const float* v = (const float*)d_in[0];   // [B][SEQ][DIM] fp32
    const float* k = (const float*)d_in[1];   // [B][SEQ][DIM] fp32

    const size_t szHalfMat = (size_t)BATCH * SEQ * DIM * sizeof(_Float16); // 32 MB
    char* ws = (char*)d_ws;
    _Float16* khT   = (_Float16*)(ws);                         // [B][DIM][SEQ] (k/T)^T
    _Float16* vhT   = (_Float16*)(ws + szHalfMat);             // [B][DIM][SEQ] v^T
    _Float16* vh    = (_Float16*)(ws + 2 * szHalfMat);         // [B][SEQ][DIM] v
    float*    S     = (float*)   (ws + 3 * szHalfMat);         // [B][DIM][DIM]
    _Float16* attnT = (_Float16*)(ws + 3 * szHalfMat
                                     + (size_t)BATCH * DIM * DIM * sizeof(float));

    float* out  = (float*)d_out;                               // [B][SEQ][DIM]
    float* attn = out + (size_t)BATCH * SEQ * DIM;             // [B][DIM][DIM]

    // 1) convert + transpose (scale folded into k)
    dim3 gcvt(SEQ / 32, DIM / 32, BATCH);
    cvt_tr_kernel<<<gcvt, 256, 0, stream>>>(k, khT, nullptr, TEMP_INV);
    cvt_tr_kernel<<<gcvt, 256, 0, stream>>>(v, vhT, vh, 1.0f);

    // 2) S = (k/T)^T v : M=N=512, K=4096; 64x64 tiles, 4 waves
    wmma_gemm_kernel<64, 64, 32, 32, 4>
        <<<dim3(DIM / 64, DIM / 64, BATCH), 128, 0, stream>>>(
            khT, vhT, S, SEQ, DIM,
            (long)DIM * SEQ, (long)DIM * SEQ, (long)DIM * DIM);

    // 3) softmax rows -> attn (fp32, d_out) + attn^T (f16, ws)
    softmax_kernel<<<dim3(DIM, BATCH), 256, 0, stream>>>(S, attn, attnT);

    // 4) out = v * attn : M=4096, N=512, K=512; 128x128 tiles, 8 waves
    wmma_gemm_kernel<128, 128, 64, 32, 8>
        <<<dim3(SEQ / 128, DIM / 128, BATCH), 256, 0, stream>>>(
            vh, attnT, out, DIM, DIM,
            (long)SEQ * DIM, (long)DIM * DIM, (long)SEQ * DIM);
}
